// BiRNN_41085657153625
// MI455X (gfx1250) — matrix operand order repaired and checked by reading, not compiled, not run on hardware
//
#include <hip/hip_runtime.h>
#include <hip/hip_bf16.h>

typedef __attribute__((ext_vector_type(8)))  float  v8f;
typedef __attribute__((ext_vector_type(16))) __bf16 v16bf;

#define B_DIM   32
#define T_SEQ   512
#define D_DIM   512
#define H_DIM   512
#define K_RNN   1024          // D + H
#define APITCH  1032          // 1024 + 8 halves pad (16B) -> bank spread
#define ZROWS   16416         // B*T + B   (outputs rows + state rows)
#define FF_DIM  2048
#define T1PITCH 2056          // 2048 + 8 halves pad

union FragBF { uint4 u[2]; v16bf v; };

// ---------------- helpers ----------------

// Branchless tanh on the hardware transcendentals:
//   tanh(x) = 1 - 2/(1 + e^{2x})
__device__ __forceinline__ float fast_tanh(float x) {
    const float LOG2E_X2 = 2.8853900817779268f;   // 2*log2(e)
    float e = __builtin_amdgcn_exp2f(x * LOG2E_X2);
    return __builtin_fmaf(-2.0f, __builtin_amdgcn_rcpf(e + 1.0f), 1.0f);
}

__device__ __forceinline__ void grid_barrier(unsigned* bar, unsigned nwg) {
    __syncthreads();
    if (threadIdx.x == 0) {
        __threadfence();   // flush z / hbuf stores to device scope
        unsigned gen = __hip_atomic_load(bar + 1, __ATOMIC_RELAXED, __HIP_MEMORY_SCOPE_AGENT);
        unsigned arrived = __hip_atomic_fetch_add(bar, 1u, __ATOMIC_ACQ_REL, __HIP_MEMORY_SCOPE_AGENT);
        if (arrived == nwg - 1u) {
            __hip_atomic_store(bar, 0u, __ATOMIC_RELAXED, __HIP_MEMORY_SCOPE_AGENT);
            __hip_atomic_fetch_add(bar + 1, 1u, __ATOMIC_RELEASE, __HIP_MEMORY_SCOPE_AGENT);
        } else {
            while (__hip_atomic_load(bar + 1, __ATOMIC_ACQUIRE, __HIP_MEMORY_SCOPE_AGENT) == gen)
                __builtin_amdgcn_s_sleep(1);
        }
    }
    __syncthreads();
}

// ---------------- conversion kernels ----------------

__global__ void cvt_kernel(const float* __restrict__ in, __bf16* __restrict__ out, size_t n) {
    size_t i = (size_t)blockIdx.x * blockDim.x + threadIdx.x;
    if (i < n) out[i] = (__bf16)in[i];
}

// in: f32 [K][N] row-major -> out: bf16 [N][K]  (N-major, K contiguous per row)
__global__ void cvt_t_kernel(const float* __restrict__ in, __bf16* __restrict__ out,
                             int K, int N) {
    size_t i = (size_t)blockIdx.x * blockDim.x + threadIdx.x;
    size_t total = (size_t)K * N;
    if (i < total) {
        int n = (int)(i / (size_t)K);
        int k = (int)(i % (size_t)K);
        out[i] = (__bf16)in[(size_t)k * N + n];
    }
}

__global__ void init_bar_kernel(unsigned* __restrict__ bar) {
    if (threadIdx.x < 16) bar[threadIdx.x] = 0u;
}

// ---------------- RNN scan (persistent, both directions) ----------------
// grid: (8, 2)  block: 128 (4 waves).  WG owns 64 H-columns; wave w owns
// N-tile w (16 cols) x BOTH M-tiles (B frag shared by the 2 WMMAs).
__global__ __launch_bounds__(128) void rnn_kernel(
    const __bf16* __restrict__ xbf,      // [B][T][D] bf16
    const int*    __restrict__ length,   // [B]
    const __bf16* __restrict__ WT,       // [2][512][1024] (Wf^T then Wb^T, contiguous)
    const float*  __restrict__ biasF,    // [512]
    const float*  __restrict__ biasB,    // [512]
    __bf16*       __restrict__ z,        // [ZROWS][1024]
    __bf16*       __restrict__ hbuf,     // [2 dir][2 parity][32][512]
    unsigned*     __restrict__ bar,      // [2 dir][8]
    unsigned nwg)
{
    extern __shared__ char smem_raw[];
    __bf16* As = (__bf16*)smem_raw;      // [32][APITCH] : cols 0..511 = x_t, 512..1023 = h

    const int dir   = blockIdx.y;
    const int tid   = threadIdx.x;
    const int lane  = tid & 31;
    const int wave  = tid >> 5;           // 0..3 -> N-tile within WG
    const int lmod  = lane & 15;
    const int lhalf = lane >> 4;
    const int cN    = blockIdx.x * 64 + wave * 16 + lmod;   // H column (0..511)

    const __bf16* Wd   = WT + (size_t)dir * (size_t)H_DIM * K_RNN;
    const float*  bias = dir ? biasB : biasF;
    unsigned*   mybar  = bar + dir * 8;
    const float biasv  = bias[cN];

    // per-row (batch) constants: tile0 rows 0..15, tile1 rows 16..31
    int Lb0[8], Lb1[8];
#pragma unroll
    for (int rr = 0; rr < 8; ++rr) {
        Lb0[rr] = length[rr + 8 * lhalf];
        Lb1[rr] = length[rr + 8 * lhalf + 16];
    }

    v8f h0 = {}, h1 = {};   // carried hidden state slices (f32)

    const int srow = tid & 31;            // staging: batch row
    const int sseg = tid >> 5;            // staging: 4 segments of 256 halves
    const int sL   = length[srow];

    for (int s = 0; s < T_SEQ; ++s) {
        __syncthreads();                  // previous step's LDS reads done
        // ---- stage A = [x_(t or rev) | h_prev] into LDS ----
        {
            uint4* dst = (uint4*)(As + (size_t)srow * APITCH + sseg * 256);
            if (sseg < 2) {
                int u = (dir == 0) ? s : ((s < sL) ? (sL - 1 - s) : s);
                const uint4* src = (const uint4*)(xbf + ((size_t)srow * T_SEQ + u) * D_DIM + sseg * 256);
#pragma unroll
                for (int i = 0; i < 32; ++i) dst[i] = src[i];
            } else if (s == 0) {
                uint4 zz = {0u, 0u, 0u, 0u};
#pragma unroll
                for (int i = 0; i < 32; ++i) dst[i] = zz;
            } else {
                const uint4* src = (const uint4*)(hbuf +
                    (((size_t)dir * 2 + ((s + 1) & 1)) * 32 + srow) * H_DIM + (sseg - 2) * 256);
#pragma unroll
                for (int i = 0; i < 32; ++i) dst[i] = src[i];
            }
        }
        __syncthreads();

        // ---- GEMM: acc = [x|h] @ W  (K = 1024 -> 32 k-tiles, 2 M-tiles/wave) ----
        v8f acc0 = {}, acc1 = {};
        const __bf16* arow0 = As + (size_t)lmod * APITCH;
        const __bf16* arow1 = As + (size_t)(lmod + 16) * APITCH;
        const __bf16* brow  = Wd + (size_t)cN * K_RNN + lhalf * 16;
#pragma unroll 4
        for (int kt = 0; kt < 32; ++kt) {
            const int k0 = kt * 32;
            FragBF fa0, fa1, fb;
            fb.u[0]  = *(const uint4*)(brow + k0);
            fb.u[1]  = *(const uint4*)(brow + k0 + 8);
            fa0.u[0] = *(const uint4*)(arow0 + k0 + lhalf * 16);
            fa0.u[1] = *(const uint4*)(arow0 + k0 + 8 + lhalf * 16);
            fa1.u[0] = *(const uint4*)(arow1 + k0 + lhalf * 16);
            fa1.u[1] = *(const uint4*)(arow1 + k0 + 8 + lhalf * 16);
            acc0 = __builtin_amdgcn_wmma_f32_16x16x32_bf16(
                false, fa0.v, false, fb.v, (short)0, acc0, false, false);
            acc1 = __builtin_amdgcn_wmma_f32_16x16x32_bf16(
                false, fa1.v, false, fb.v, (short)0, acc1, false, false);
        }

        // ---- epilogue: tanh, mask, carry h, emit z & h slices ----
#pragma unroll
        for (int rr = 0; rr < 8; ++rr) {
            const int b = rr + 8 * lhalf;           // tile 0: batches 0..15
            const bool m = (s < Lb0[rr]);
            const float hn = fast_tanh(acc0[rr] + biasv);
            const float ov = m ? hn : 0.0f;
            h0[rr] = m ? hn : h0[rr];
            const int u = (dir == 0) ? s : (m ? (Lb0[rr] - 1 - s) : s);
            z[((size_t)b * T_SEQ + u) * 1024 + dir * H_DIM + cN] = (__bf16)ov;
            hbuf[(((size_t)dir * 2 + (s & 1)) * 32 + b) * H_DIM + cN] = (__bf16)h0[rr];
        }
#pragma unroll
        for (int rr = 0; rr < 8; ++rr) {
            const int b = rr + 8 * lhalf + 16;      // tile 1: batches 16..31
            const bool m = (s < Lb1[rr]);
            const float hn = fast_tanh(acc1[rr] + biasv);
            const float ov = m ? hn : 0.0f;
            h1[rr] = m ? hn : h1[rr];
            const int u = (dir == 0) ? s : (m ? (Lb1[rr] - 1 - s) : s);
            z[((size_t)b * T_SEQ + u) * 1024 + dir * H_DIM + cN] = (__bf16)ov;
            hbuf[(((size_t)dir * 2 + (s & 1)) * 32 + b) * H_DIM + cN] = (__bf16)h1[rr];
        }
        grid_barrier(mybar, nwg);
    }

    // final states -> z rows 16384..16415
#pragma unroll
    for (int rr = 0; rr < 8; ++rr) {
        const int b0 = rr + 8 * lhalf;
        z[((size_t)(B_DIM * T_SEQ) + b0) * 1024 + dir * H_DIM + cN] = (__bf16)h0[rr];
        z[((size_t)(B_DIM * T_SEQ) + b0 + 16) * 1024 + dir * H_DIM + cN] = (__bf16)h1[rr];
    }
}

// ---------------- fused FFN: out = tanh(z@W1+b1)@W2+b2 ----------------
// grid: 513 (32 rows each), block: 256 (8 waves). Intermediate in LDS.
// 2x2 register blocking: per k-tile, 2 A-frags + 2 B-frags feed 4 WMMAs.
__global__ __launch_bounds__(256) void ffn_kernel(
    const __bf16* __restrict__ z,     // [ZROWS][1024]
    const __bf16* __restrict__ W1T,   // [2048][1024]
    const float*  __restrict__ b1,    // [2048]
    const __bf16* __restrict__ W2T,   // [1024][2048]
    const float*  __restrict__ b2,    // [1024]
    float*        __restrict__ out)   // [ZROWS][1024]
{
    extern __shared__ char smem_raw[];
    __bf16* Zs = (__bf16*)smem_raw;                             // [32][APITCH]
    __bf16* T1 = (__bf16*)(smem_raw + 32 * APITCH * 2);         // [32][T1PITCH]

    const int tid   = threadIdx.x;
    const int lane  = tid & 31;
    const int wave  = tid >> 5;         // 0..7
    const int lmod  = lane & 15;
    const int lhalf = lane >> 4;
    const size_t row0 = (size_t)blockIdx.x * 32;

    // stage 32 z-rows
    {
        const int r = tid & 31, seg = tid >> 5;     // 8 segs x 128 halves
        const uint4* src = (const uint4*)(z + (row0 + r) * 1024 + seg * 128);
        uint4* dst = (uint4*)(Zs + (size_t)r * APITCH + seg * 128);
#pragma unroll
        for (int i = 0; i < 16; ++i) dst[i] = src[i];
    }
    __syncthreads();

    const __bf16* arow0z = Zs + (size_t)lmod * APITCH;
    const __bf16* arow1z = Zs + (size_t)(lmod + 16) * APITCH;

    // phase 1: T1 = tanh(Z @ W1 + b1)  (wave: 8 N-tile pairs x 2 M-tiles)
    for (int j = 0; j < 8; ++j) {
        const int nA = (wave * 16 + 2 * j) * 16 + lmod;     // 0..2047
        const int nB = nA + 16;
        const float bvA = b1[nA], bvB = b1[nB];
        const __bf16* browA = W1T + (size_t)nA * 1024 + lhalf * 16;
        const __bf16* browB = W1T + (size_t)nB * 1024 + lhalf * 16;
        if (j + 1 < 8) __builtin_prefetch(W1T + (size_t)(nA + 32) * 1024, 0, 1);
        v8f a00 = {}, a01 = {}, a10 = {}, a11 = {};
#pragma unroll 4
        for (int kt = 0; kt < 32; ++kt) {
            const int k0 = kt * 32;
            FragBF fa0, fa1, fbA, fbB;
            fbA.u[0] = *(const uint4*)(browA + k0);
            fbA.u[1] = *(const uint4*)(browA + k0 + 8);
            fbB.u[0] = *(const uint4*)(browB + k0);
            fbB.u[1] = *(const uint4*)(browB + k0 + 8);
            fa0.u[0] = *(const uint4*)(arow0z + k0 + lhalf * 16);
            fa0.u[1] = *(const uint4*)(arow0z + k0 + 8 + lhalf * 16);
            fa1.u[0] = *(const uint4*)(arow1z + k0 + lhalf * 16);
            fa1.u[1] = *(const uint4*)(arow1z + k0 + 8 + lhalf * 16);
            a00 = __builtin_amdgcn_wmma_f32_16x16x32_bf16(false, fa0.v, false, fbA.v, (short)0, a00, false, false);
            a01 = __builtin_amdgcn_wmma_f32_16x16x32_bf16(false, fa0.v, false, fbB.v, (short)0, a01, false, false);
            a10 = __builtin_amdgcn_wmma_f32_16x16x32_bf16(false, fa1.v, false, fbA.v, (short)0, a10, false, false);
            a11 = __builtin_amdgcn_wmma_f32_16x16x32_bf16(false, fa1.v, false, fbB.v, (short)0, a11, false, false);
        }
#pragma unroll
        for (int rr = 0; rr < 8; ++rr) {
            const int m0 = rr + 8 * lhalf;
            T1[(size_t)m0 * T1PITCH + nA]        = (__bf16)fast_tanh(a00[rr] + bvA);
            T1[(size_t)m0 * T1PITCH + nB]        = (__bf16)fast_tanh(a01[rr] + bvB);
            T1[(size_t)(m0 + 16) * T1PITCH + nA] = (__bf16)fast_tanh(a10[rr] + bvA);
            T1[(size_t)(m0 + 16) * T1PITCH + nB] = (__bf16)fast_tanh(a11[rr] + bvB);
        }
    }
    __syncthreads();

    const __bf16* arow0t = T1 + (size_t)lmod * T1PITCH;
    const __bf16* arow1t = T1 + (size_t)(lmod + 16) * T1PITCH;

    // phase 2: out = T1 @ W2 + b2  (wave: 4 N-tile pairs x 2 M-tiles, K=2048)
    for (int j = 0; j < 4; ++j) {
        const int nA = (wave * 8 + 2 * j) * 16 + lmod;      // 0..1023
        const int nB = nA + 16;
        const float bvA = b2[nA], bvB = b2[nB];
        const __bf16* browA = W2T + (size_t)nA * FF_DIM + lhalf * 16;
        const __bf16* browB = W2T + (size_t)nB * FF_DIM + lhalf * 16;
        if (j + 1 < 4) __builtin_prefetch(W2T + (size_t)(nA + 32) * FF_DIM, 0, 1);
        v8f a00 = {}, a01 = {}, a10 = {}, a11 = {};
#pragma unroll 4
        for (int kt = 0; kt < 64; ++kt) {
            const int k0 = kt * 32;
            FragBF fa0, fa1, fbA, fbB;
            fbA.u[0] = *(const uint4*)(browA + k0);
            fbA.u[1] = *(const uint4*)(browA + k0 + 8);
            fbB.u[0] = *(const uint4*)(browB + k0);
            fbB.u[1] = *(const uint4*)(browB + k0 + 8);
            fa0.u[0] = *(const uint4*)(arow0t + k0 + lhalf * 16);
            fa0.u[1] = *(const uint4*)(arow0t + k0 + 8 + lhalf * 16);
            fa1.u[0] = *(const uint4*)(arow1t + k0 + lhalf * 16);
            fa1.u[1] = *(const uint4*)(arow1t + k0 + 8 + lhalf * 16);
            a00 = __builtin_amdgcn_wmma_f32_16x16x32_bf16(false, fa0.v, false, fbA.v, (short)0, a00, false, false);
            a01 = __builtin_amdgcn_wmma_f32_16x16x32_bf16(false, fa0.v, false, fbB.v, (short)0, a01, false, false);
            a10 = __builtin_amdgcn_wmma_f32_16x16x32_bf16(false, fa1.v, false, fbA.v, (short)0, a10, false, false);
            a11 = __builtin_amdgcn_wmma_f32_16x16x32_bf16(false, fa1.v, false, fbB.v, (short)0, a11, false, false);
        }
#pragma unroll
        for (int rr = 0; rr < 8; ++rr) {
            const int m0 = rr + 8 * lhalf;
            out[(row0 + m0) * 1024 + nA]        = a00[rr] + bvA;
            out[(row0 + m0) * 1024 + nB]        = a01[rr] + bvB;
            out[(row0 + m0 + 16) * 1024 + nA]   = a10[rr] + bvA;
            out[(row0 + m0 + 16) * 1024 + nB]   = a11[rr] + bvB;
        }
    }
}

// ---------------- workspace layout (bytes) ----------------
static const size_t OFF_XBF = 0;                          // 16,777,216
static const size_t OFF_WT  = OFF_XBF + 16777216;         //  2 x 1,048,576 (WfT, WbT contiguous)
static const size_t OFF_W1T = OFF_WT  + 2097152;          //  4,194,304
static const size_t OFF_W2T = OFF_W1T + 4194304;          //  4,194,304
static const size_t OFF_Z   = OFF_W2T + 4194304;          // 33,619,968
static const size_t OFF_H   = OFF_Z   + (size_t)ZROWS * 1024 * 2;  // 131,072
static const size_t OFF_BAR = OFF_H   + 131072;           // 64

extern "C" void kernel_launch(void* const* d_in, const int* in_sizes, int n_in,
                              void* d_out, int out_size, void* d_ws, size_t ws_size,
                              hipStream_t stream) {
    const float* x    = (const float*)d_in[0];
    const int*   len  = (const int*)  d_in[1];
    const float* Wf   = (const float*)d_in[2];
    const float* bf_  = (const float*)d_in[3];
    const float* Wb   = (const float*)d_in[4];
    const float* bb_  = (const float*)d_in[5];
    const float* W1   = (const float*)d_in[6];
    const float* b1   = (const float*)d_in[7];
    const float* W2   = (const float*)d_in[8];
    const float* b2   = (const float*)d_in[9];

    char* ws = (char*)d_ws;
    __bf16*   xbf  = (__bf16*)(ws + OFF_XBF);
    __bf16*   WfT  = (__bf16*)(ws + OFF_WT);                 // WbT directly after
    __bf16*   WbT  = WfT + (size_t)H_DIM * K_RNN;
    __bf16*   W1T  = (__bf16*)(ws + OFF_W1T);
    __bf16*   W2T  = (__bf16*)(ws + OFF_W2T);
    __bf16*   zbuf = (__bf16*)(ws + OFF_Z);
    __bf16*   hbuf = (__bf16*)(ws + OFF_H);
    unsigned* bar  = (unsigned*)(ws + OFF_BAR);

    const size_t rnn_lds = (size_t)32 * APITCH * 2;                         // 66,048 B
    const size_t ffn_lds = (size_t)32 * APITCH * 2 + (size_t)32 * T1PITCH * 2; // 197,632 B
    (void)hipFuncSetAttribute((const void*)rnn_kernel,
        hipFuncAttributeMaxDynamicSharedMemorySize, (int)rnn_lds);
    (void)hipFuncSetAttribute((const void*)ffn_kernel,
        hipFuncAttributeMaxDynamicSharedMemorySize, (int)ffn_lds);

    // f32 -> bf16 conversions / weight transposes
    {
        const size_t nx = (size_t)B_DIM * T_SEQ * D_DIM;
        cvt_kernel<<<(unsigned)((nx + 255) / 256), 256, 0, stream>>>(x, xbf, nx);
        cvt_t_kernel<<<(unsigned)((524288 + 255) / 256), 256, 0, stream>>>(Wf, WfT, K_RNN, H_DIM);
        cvt_t_kernel<<<(unsigned)((524288 + 255) / 256), 256, 0, stream>>>(Wb, WbT, K_RNN, H_DIM);
        cvt_t_kernel<<<(unsigned)((2097152 + 255) / 256), 256, 0, stream>>>(W1, W1T, 1024, FF_DIM);
        cvt_t_kernel<<<(unsigned)((2097152 + 255) / 256), 256, 0, stream>>>(W2, W2T, FF_DIM, 1024);
    }
    init_bar_kernel<<<1, 32, 0, stream>>>(bar);

    // persistent bidirectional scan: 8 WGs per direction, grid barrier per step
    dim3 grnn(8, 2);
    rnn_kernel<<<grnn, 128, rnn_lds, stream>>>(xbf, len, WfT, bf_, bb_, zbuf, hbuf, bar, 8u);

    // fused FFN over 16416 rows (outputs + states) -> d_out
    ffn_kernel<<<ZROWS / 32, 256, ffn_lds, stream>>>(zbuf, W1T, b1, W2T, b2, (float*)d_out);
}
